// WindowSelfAttention_80393197846723
// MI455X (gfx1250) — matrix operand order repaired and checked by reading, not compiled, not run on hardware
//
#include <hip/hip_runtime.h>
#include <hip/hip_bf16.h>

#define HEADS 12
#define EMBED 384
#define HDIM  32
#define NTOK  49
#define NPAD  64
#define NWIN  64
#define XS    392   // f16 row stride for sX/sQ/sK (64 rows x 384 cols, padded)
#define VTS   72    // f16 row stride for sVt (384 rows x 64 cols, padded)
#define SSTR  68    // f32 row stride for score buffer
#define PSTR  72    // f16 row stride for probability buffer
#define MSTR  52    // f32 row stride for staged mask window

typedef __attribute__((ext_vector_type(16))) _Float16 v16h;
typedef __attribute__((ext_vector_type(8)))  _Float16 v8h;
typedef __attribute__((ext_vector_type(4)))  _Float16 v4h;
typedef __attribute__((ext_vector_type(8)))  float    v8f;

// A-fragment (16x32 f16, M x K): lane L<16 -> row M=L, K 0..7 (lo) and 16..23 (hi);
// lane L+16 -> row M=L, K 8..15 (lo) and 24..31 (hi).  (ISA 7.12.2)
static __device__ inline v16h load_fragA(const _Float16* buf, int stride,
                                         int mbase, int kbase, int lane) {
  int m  = mbase + (lane & 15);
  int k0 = kbase + ((lane >> 4) << 3);
  const _Float16* p = buf + m * stride + k0;
  v8h lo = *(const v8h*)(p);
  v8h hi = *(const v8h*)(p + 16);
  v16h a;
#pragma unroll
  for (int i = 0; i < 8; ++i) { a[i] = lo[i]; a[i + 8] = hi[i]; }
  return a;
}

// B-fragment (32x16 f16, K x N), buf holds B^T rows: buf[n][k] contiguous in k.
// lanes 0..15 -> K 0..7 (lo) and 16..23 (hi); lanes 16..31 -> K 8..15 (lo) and 24..31 (hi).
static __device__ inline v16h load_fragB(const _Float16* buf, int stride,
                                         int nbase, int kbase, int lane) {
  int n  = nbase + (lane & 15);
  int k0 = kbase + ((lane >> 4) << 3);
  const _Float16* p = buf + n * stride + k0;
  v8h lo = *(const v8h*)(p);
  v8h hi = *(const v8h*)(p + 16);
  v16h b;
#pragma unroll
  for (int i = 0; i < 8; ++i) { b[i] = lo[i]; b[i + 8] = hi[i]; }
  return b;
}

static __device__ inline v8f wmma_f16(v16h a, v16h b, v8f c) {
  return __builtin_amdgcn_wmma_f32_16x16x32_f16(false, a, false, b,
                                                (short)0, c, false, false);
}

__global__ void convert_weights_kernel(const float* __restrict__ Wq,
                                       const float* __restrict__ Wk,
                                       const float* __restrict__ Wv,
                                       const float* __restrict__ Wo,
                                       _Float16* __restrict__ o) {
  int i = blockIdx.x * 256 + threadIdx.x;
  const int S = EMBED * EMBED;
  if (i >= 4 * S) return;
  const float* src = (i < S) ? Wq : (i < 2 * S) ? Wk : (i < 3 * S) ? Wv : Wo;
  o[i] = (_Float16)src[i % S];
}

__global__ __launch_bounds__(256, 1) void win_attn_kernel(
    const float* __restrict__ query, const float* __restrict__ key_,
    const float* __restrict__ value, const float* __restrict__ mask,
    const float* __restrict__ bq, const float* __restrict__ bk,
    const float* __restrict__ bv, const float* __restrict__ bo,
    const float* __restrict__ rel_pos,
    const _Float16* __restrict__ W16,   // [4][384*384] f16: Wq,Wk,Wv,Wo (row-major [e][c])
    float* __restrict__ out) {
  __shared__ _Float16 sX[NPAD * XS];    // input window / later attention output O
  __shared__ _Float16 sQ[NPAD * XS];
  __shared__ _Float16 sK[NPAD * XS];
  __shared__ _Float16 sVt[EMBED * VTS]; // V transposed: [d][token]
  __shared__ float    sS[NPAD * SSTR];
  __shared__ _Float16 sP[NPAD * PSTR];
  __shared__ float    sM[NPAD * MSTR];          // staged mask window [49][49]
  __shared__ float    sRel[13 * 13 * HEADS];    // staged rel_pos table

  const int b    = blockIdx.x;
  const int tid  = threadIdx.x;
  const int lane = tid & 31;
  const int wave = tid >> 5;

  const size_t wbase = (size_t)b * NTOK * EMBED;
  const float* maskw = mask + (size_t)(b & (NWIN - 1)) * NTOK * NTOK;

  // one-time staging: zero pad rows of sX, mask window, rel_pos table
  for (int i = tid; i < (NPAD - NTOK) * EMBED; i += 256)
    sX[(NTOK + i / EMBED) * XS + (i % EMBED)] = (_Float16)0.f;
  for (int i = tid; i < NTOK * NTOK; i += 256)
    sM[(i / NTOK) * MSTR + (i % NTOK)] = maskw[i];
  for (int i = tid; i < 13 * 13 * HEADS; i += 256)
    sRel[i] = rel_pos[i];

  // ---------------- Q/K/V projections: out = X @ W^T + b ----------------
  // Register-blocked: each wave owns N-panels (nt = wave, wave+8, wave+16),
  // reuses one B-fragment across 4 M-tiles per k-step (4 accumulators).
  const float* inputs[3] = {query, key_, value};
  const float* biases[3] = {bq, bk, bv};
#pragma unroll 1
  for (int p = 0; p < 3; ++p) {
    const float4* src4 = (const float4*)(inputs[p] + wbase);
    for (int i = tid; i < NTOK * EMBED / 4; i += 256) {
      float4 v = src4[i];
      int row = i / (EMBED / 4), c4 = (i % (EMBED / 4)) * 4;
      v4h h4 = { (_Float16)v.x, (_Float16)v.y, (_Float16)v.z, (_Float16)v.w };
      *(v4h*)(sX + row * XS + c4) = h4;
    }
    __syncthreads();
    const _Float16* W = W16 + (size_t)p * EMBED * EMBED;
    const float* bias = biases[p];
#pragma unroll 1
    for (int nt = wave; nt < EMBED / 16; nt += 8) {
      if (nt + 8 < EMBED / 16)  // prefetch next weight panel into caches
        __builtin_prefetch(W + (size_t)(nt + 8) * 16 * EMBED + lane * 192, 0, 1);
      v8f acc[4] = {{}, {}, {}, {}};
#pragma unroll 1
      for (int kk = 0; kk < EMBED / 32; ++kk) {
        v16h bf = load_fragB(W, EMBED, nt * 16, kk * 32, lane); // B[k][n] = W[n][k]
#pragma unroll
        for (int mt = 0; mt < 4; ++mt) {
          v16h a = load_fragA(sX, XS, mt * 16, kk * 32, lane);
          acc[mt] = wmma_f16(a, bf, acc[mt]);
        }
      }
      int n = nt * 16 + (lane & 15);
      float bval = bias[n];
#pragma unroll
      for (int mt = 0; mt < 4; ++mt) {
        int m0 = mt * 16 + ((lane >> 4) << 3);
        if (p == 0) {
#pragma unroll
          for (int r = 0; r < 8; ++r) sQ[(m0 + r) * XS + n] = (_Float16)(acc[mt][r] + bval);
        } else if (p == 1) {
#pragma unroll
          for (int r = 0; r < 8; ++r) sK[(m0 + r) * XS + n] = (_Float16)(acc[mt][r] + bval);
        } else {
#pragma unroll
          for (int r = 0; r < 8; ++r) sVt[n * VTS + (m0 + r)] = (_Float16)(acc[mt][r] + bval);
        }
      }
    }
    __syncthreads();
  }

  // ---------------- per-head attention ----------------
  const float scale = 0.17677669529663687f; // 1/sqrt(32)
#pragma unroll 1
  for (int h = 0; h < HEADS; ++h) {
    // S = Q_h K_h^T * scale + rel_bias + mask  (16 tiles; each wave: 1 A-frag, 2 tiles)
    {
      int mt = wave >> 1;
      v16h a = load_fragA(sQ, XS, mt * 16, h * HDIM, lane);
#pragma unroll
      for (int j = 0; j < 2; ++j) {
        int nt = (wave & 1) * 2 + j;
        v16h bf = load_fragB(sK, XS, nt * 16, h * HDIM, lane); // B[k][n] = K[n][h*32+k]
        v8f acc = {};
        acc = wmma_f16(a, bf, acc);
        int col = nt * 16 + (lane & 15);
        int m0  = mt * 16 + ((lane >> 4) << 3);
        int kx = 0, ky = 0;
        if (col < NTOK) { kx = col / 7; ky = col - kx * 7; }
#pragma unroll
        for (int r = 0; r < 8; ++r) {
          int row = m0 + r;
          float v;
          if (col < NTOK) {
            float add = 0.f;
            if (row < NTOK) {
              int qx = row / 7, qy = row - qx * 7;
              int ri = qx - kx; ri = (ri < 0) ? ri + 13 : ri;  // numpy mod wrap
              int rj = qy - ky; rj = (rj < 0) ? rj + 13 : rj;
              add = sRel[(ri * 13 + rj) * HEADS + h] + sM[row * MSTR + col];
            }
            v = acc[r] * scale + add;
          } else {
            v = -3.0e38f;  // padded key columns -> softmax weight 0
          }
          sS[row * SSTR + col] = v;
        }
      }
    }
    __syncthreads();

    // softmax: 4 lanes per row (same wave), registers + shuffle reductions
    {
      int row = tid >> 2;
      int sub = tid & 3;
      const float* sr = sS + row * SSTR + sub * 16;
      float e[16];
      float mx = -3.0e38f;
#pragma unroll
      for (int c = 0; c < 16; ++c) { e[c] = sr[c]; mx = fmaxf(mx, e[c]); }
      mx = fmaxf(mx, __shfl_xor(mx, 1, 32));
      mx = fmaxf(mx, __shfl_xor(mx, 2, 32));
      float sum = 0.f;
#pragma unroll
      for (int c = 0; c < 16; ++c) { e[c] = __expf(e[c] - mx); sum += e[c]; }
      sum += __shfl_xor(sum, 1, 32);
      sum += __shfl_xor(sum, 2, 32);
      float inv = 1.f / sum;  // >= 1 guaranteed (max element contributes exp(0)=1)
      _Float16* pr = sP + row * PSTR + sub * 16;
#pragma unroll
      for (int c = 0; c < 16; ++c) pr[c] = (_Float16)(e[c] * inv);
    }
    __syncthreads();

    // O_h = P @ V_h   (8 tiles, 1 per wave, K=64 in 2 steps)
    {
      int mt = wave >> 1, nt = wave & 1;
      v8f acc = {};
#pragma unroll
      for (int kk = 0; kk < 2; ++kk) {
        v16h a  = load_fragA(sP, PSTR, mt * 16, kk * 32, lane);
        v16h bf = load_fragB(sVt, VTS, h * HDIM + nt * 16, kk * 32, lane); // B[k][n]=Vt[d][k]
        acc = wmma_f16(a, bf, acc);
      }
      int n  = h * HDIM + nt * 16 + (lane & 15);
      int m0 = mt * 16 + ((lane >> 4) << 3);
#pragma unroll
      for (int r = 0; r < 8; ++r) sX[(m0 + r) * XS + n] = (_Float16)acc[r];
    }
    __syncthreads();
  }

  // ---------------- output projection: out = O @ Wo^T + bo ----------------
  const _Float16* Wo16 = W16 + (size_t)3 * EMBED * EMBED;
  float* outp = out + wbase;
#pragma unroll 1
  for (int nt = wave; nt < EMBED / 16; nt += 8) {
    if (nt + 8 < EMBED / 16)
      __builtin_prefetch(Wo16 + (size_t)(nt + 8) * 16 * EMBED + lane * 192, 0, 1);
    v8f acc[4] = {{}, {}, {}, {}};
#pragma unroll 1
    for (int kk = 0; kk < EMBED / 32; ++kk) {
      v16h bf = load_fragB(Wo16, EMBED, nt * 16, kk * 32, lane);
#pragma unroll
      for (int mt = 0; mt < 4; ++mt) {
        v16h a = load_fragA(sX, XS, mt * 16, kk * 32, lane);
        acc[mt] = wmma_f16(a, bf, acc[mt]);
      }
    }
    int n = nt * 16 + (lane & 15);
    float bval = bo[n];
#pragma unroll
    for (int mt = 0; mt < 4; ++mt) {
      int m0 = mt * 16 + ((lane >> 4) << 3);
#pragma unroll
      for (int r = 0; r < 8; ++r) {
        int row = m0 + r;
        if (row < NTOK) outp[row * EMBED + n] = acc[mt][r] + bval;
      }
    }
  }
}

extern "C" void kernel_launch(void* const* d_in, const int* in_sizes, int n_in,
                              void* d_out, int out_size, void* d_ws, size_t ws_size,
                              hipStream_t stream) {
  const float* query   = (const float*)d_in[0];
  const float* key_    = (const float*)d_in[1];
  const float* value   = (const float*)d_in[2];
  const float* mask    = (const float*)d_in[3];
  const float* Wq      = (const float*)d_in[4];
  const float* bq      = (const float*)d_in[5];
  const float* Wk      = (const float*)d_in[6];
  const float* bk      = (const float*)d_in[7];
  const float* Wv      = (const float*)d_in[8];
  const float* bv      = (const float*)d_in[9];
  const float* Wo      = (const float*)d_in[10];
  const float* bo      = (const float*)d_in[11];
  const float* rel_pos = (const float*)d_in[12];

  _Float16* W16 = (_Float16*)d_ws;  // 4 * 384*384 f16 = 1.18 MB

  int totalW = 4 * EMBED * EMBED;
  convert_weights_kernel<<<(totalW + 255) / 256, 256, 0, stream>>>(Wq, Wk, Wv, Wo, W16);

  int nblocks = in_sizes[0] / (NTOK * EMBED);  // 4096 windows
  win_attn_kernel<<<nblocks, 256, 0, stream>>>(query, key_, value, mask,
                                               bq, bk, bv, bo, rel_pos, W16,
                                               (float*)d_out);
}
